// Attention_13194139533584
// MI455X (gfx1250) — matrix operand order repaired and checked by reading, not compiled, not run on hardware
//
#include <hip/hip_runtime.h>
#include <hip/hip_bf16.h>

// ---------------------------------------------------------------------------
// Types for CDNA5 WMMA (wave32): A/B = v16h (16 x f16 per lane), C/D = v8f.
// ---------------------------------------------------------------------------
typedef __attribute__((ext_vector_type(16))) _Float16 v16h;
typedef __attribute__((ext_vector_type(8)))  _Float16 v8h;
typedef __attribute__((ext_vector_type(8)))  float    v8f;

#define S_LEN   3072
#define DMODEL  4096
#define NH      32
#define NKV     8
#define HD      128
#define KCACHE  2048
#define WINDOW  1024

__device__ __forceinline__ v16h cat8(v8h lo, v8h hi) {
  return __builtin_shufflevector(lo, hi, 0,1,2,3,4,5,6,7,8,9,10,11,12,13,14,15);
}

// A-fragment (16x32 f16, MxK): lanes 0-15 -> M=lane, K 0..15;
// lanes 16-31 -> M=lane-16, K 16..31. rowPtr points at this lane's row,
// at the start of the current 32-wide K chunk.
__device__ __forceinline__ v16h load_a_frag(const _Float16* rowPtr, int halfSel) {
  v8h lo = *(const v8h*)(rowPtr + halfSel * 16);
  v8h hi = *(const v8h*)(rowPtr + halfSel * 16 + 8);
  return cat8(lo, hi);
}

// ---------------------------------------------------------------------------
// f32 -> f16 convert
// ---------------------------------------------------------------------------
__global__ void f32_to_f16(const float* __restrict__ src, _Float16* __restrict__ dst, int n) {
  int i = blockIdx.x * 256 + threadIdx.x;
  if (i < n) dst[i] = (_Float16)src[i];
}

// ---------------------------------------------------------------------------
// GEMM: C[M,N] (f32) = A[M,K] (f16, row-major) * B[N,K]^T (f16, row-major)
// Block: 256 threads = 8 waves (4 along M x 2 along N), tile 128x128.
// Each wave: 2 (M) x 4 (N) accumulators of 16x16, K stepped by 32.
// Fragments loaded directly from global (L2-resident weights on MI455X).
// ---------------------------------------------------------------------------
__global__ __launch_bounds__(256)
void gemm_nt_f16(const _Float16* __restrict__ A, const _Float16* __restrict__ B,
                 float* __restrict__ C, int M, int N, int K) {
  const int tid    = threadIdx.x;
  const int lane   = tid & 31;
  const int wave   = tid >> 5;
  const int waveM  = wave & 3;
  const int waveN  = wave >> 2;
  const int lanelo = lane & 15;
  const int halfSel = lane >> 4;

  const int mBase = blockIdx.x * 128 + waveM * 32;
  const int nBase = blockIdx.y * 128 + waveN * 64;

  v8f acc[2][4];
#pragma unroll
  for (int mt = 0; mt < 2; ++mt)
#pragma unroll
    for (int nt = 0; nt < 4; ++nt) acc[mt][nt] = v8f{};

  const _Float16* aRow0 = A + (size_t)(mBase + lanelo) * K;
  const _Float16* aRow1 = A + (size_t)(mBase + 16 + lanelo) * K;
  const _Float16* bRow[4];
#pragma unroll
  for (int nt = 0; nt < 4; ++nt)
    bRow[nt] = B + (size_t)(nBase + nt * 16 + lanelo) * K;

  for (int k0 = 0; k0 < K; k0 += 32) {
    v16h a0 = load_a_frag(aRow0 + k0, halfSel);
    v16h a1 = load_a_frag(aRow1 + k0, halfSel);
#pragma unroll
    for (int nt = 0; nt < 4; ++nt) {
      // B-fragment (32x16, KxN): lanes 0-15 K=0..15 (N=lane), lanes 16-31 K=16..31.
      v16h b = *(const v16h*)(bRow[nt] + k0 + halfSel * 16);
      acc[0][nt] = __builtin_amdgcn_wmma_f32_16x16x32_f16(false, a0, false, b,
                                                          (short)0, acc[0][nt], false, false);
      acc[1][nt] = __builtin_amdgcn_wmma_f32_16x16x32_f16(false, a1, false, b,
                                                          (short)0, acc[1][nt], false, false);
    }
  }

  // C/D layout: VGPR r, lanes 0-15 -> row r, N=lane; lanes 16-31 -> row r+8.
#pragma unroll
  for (int mt = 0; mt < 2; ++mt)
#pragma unroll
    for (int nt = 0; nt < 4; ++nt)
#pragma unroll
      for (int r = 0; r < 8; ++r) {
        int row = mBase + mt * 16 + r + halfSel * 8;
        int col = nBase + nt * 16 + lanelo;
        C[(size_t)row * N + col] = acc[mt][nt][r];
      }
}

// ---------------------------------------------------------------------------
// RoPE + re-layout to head-major f16: src [S, nheads*128] f32 -> dst [nheads][S][128] f16
// ---------------------------------------------------------------------------
__global__ void rope_to_heads(const float* __restrict__ src, const float* __restrict__ freqs,
                              _Float16* __restrict__ dst, int nheads) {
  int i = blockIdx.x * 256 + threadIdx.x;        // over S * nheads * 64 pairs
  int total = S_LEN * nheads * 64;
  if (i >= total) return;
  int p = i & 63;
  int h = (i >> 6) % nheads;
  int s = i / (64 * nheads);
  float ang = freqs[s * 64 + p];
  float c = cosf(ang), sn = sinf(ang);
  const float* base = src + (size_t)s * (nheads * HD) + h * HD + 2 * p;
  float a = base[0], b = base[1];
  _Float16* out = dst + ((size_t)h * S_LEN + s) * HD + 2 * p;
  out[0] = (_Float16)(a * c - b * sn);
  out[1] = (_Float16)(a * sn + b * c);
}

// V: f32 [S, 8*128] -> f16 head-major [8][S][128]
__global__ void v_to_heads(const float* __restrict__ src, _Float16* __restrict__ dst) {
  int i = blockIdx.x * 256 + threadIdx.x;        // over S * 1024
  if (i >= S_LEN * NKV * HD) return;
  int d = i & 127;
  int h = (i >> 7) & 7;
  int s = i >> 10;
  dst[((size_t)h * S_LEN + s) * HD + d] = (_Float16)src[(size_t)s * (NKV * HD) + h * HD + d];
}

// ---------------------------------------------------------------------------
// Flash attention over the sliding-window cache.
// Cache key index ks in [0,2048): key position = ks + (ks >= 1024 ? 1024 : 0).
// Grid: (S/64, 32 heads). Block: 128 threads = 4 waves; wave w owns q rows
// [w*16, w*16+16). Q and K fragments come straight from global (d-contiguous);
// V is transposed through LDS for PV B-fragments; P goes through LDS to be
// re-read in A-fragment layout.
// ---------------------------------------------------------------------------
#define QT 64
#define KT 64

__global__ __launch_bounds__(128)
void flash_attn(const _Float16* __restrict__ qh, const _Float16* __restrict__ kh,
                const _Float16* __restrict__ vh, _Float16* __restrict__ oh) {
  __shared__ _Float16 VT[HD][KT + 8];   // [d][key], transposed V tile
  __shared__ _Float16 Pt[QT][KT + 8];   // [q][key], probabilities

  const int qb   = blockIdx.x * QT;
  const int h    = blockIdx.y;
  const int hkv  = h >> 2;              // GQA: 4 Q heads per KV head
  const int tid  = threadIdx.x;
  const int lane = tid & 31;
  const int wave = tid >> 5;            // 0..3
  const int lanelo  = lane & 15;
  const int halfSel = lane >> 4;
  const int m0 = wave * 16;             // this wave's local q-row base

  // Preload this wave's Q fragments: 4 K-chunks of 32 covering HD=128.
  const _Float16* qrow = qh + ((size_t)h * S_LEN + (qb + m0 + lanelo)) * HD;
  v16h qf[4];
#pragma unroll
  for (int kk = 0; kk < 4; ++kk) qf[kk] = load_a_frag(qrow + kk * 32, halfSel);

  v8f oacc[8];
#pragma unroll
  for (int dt = 0; dt < 8; ++dt) oacc[dt] = v8f{};
  float mrow[8], lrow[8];
#pragma unroll
  for (int r = 0; r < 8; ++r) { mrow[r] = -1e30f; lrow[r] = 0.0f; }

  const float scale = 0.08838834764831845f;   // 1/sqrt(128)

  for (int kt0 = 0; kt0 < KCACHE; kt0 += KT) {
    const int posBase = kt0 + (kt0 >= WINDOW ? WINDOW : 0);
    if (posBase > qb + QT - 1) break;         // positions monotone in kt0: uniform exit

    // Stage V tile transposed: keys [kt0, kt0+64) x d [0,128) -> VT[d][key]
#pragma unroll
    for (int i = 0; i < 8; ++i) {
      int c   = tid + i * 128;                // 1024 chunks of 8 f16
      int key = c >> 4;
      int dc  = (c & 15) * 8;
      const _Float16* vptr = vh + ((size_t)hkv * S_LEN + (kt0 + key)) * HD + dc;
      v8h v = *(const v8h*)vptr;
#pragma unroll
      for (int j = 0; j < 8; ++j) VT[dc + j][key] = v[j];
    }
    __syncthreads();

    // Scores: S = Q * K^T for 4 n-tiles of 16 keys, scaled + masked.
    float mtile[8];
#pragma unroll
    for (int r = 0; r < 8; ++r) mtile[r] = -1e30f;
    v8f sacc[4];
#pragma unroll
    for (int nt = 0; nt < 4; ++nt) {
      v8f acc = v8f{};
      const _Float16* krow = kh + ((size_t)hkv * S_LEN + (kt0 + nt * 16 + lanelo)) * HD;
#pragma unroll
      for (int kk = 0; kk < 4; ++kk) {
        v16h b = *(const v16h*)(krow + kk * 32 + halfSel * 16);
        acc = __builtin_amdgcn_wmma_f32_16x16x32_f16(false, qf[kk], false, b,
                                                     (short)0, acc, false, false);
      }
#pragma unroll
      for (int r = 0; r < 8; ++r) {
        int qpos = qb + m0 + r + halfSel * 8;
        int ks   = kt0 + nt * 16 + lanelo;
        int kpos = ks + (ks >= WINDOW ? WINDOW : 0);
        float v  = acc[r] * scale;
        v = (qpos >= kpos) ? v : -1e30f;
        acc[r] = v;
        mtile[r] = fmaxf(mtile[r], v);
      }
      sacc[nt] = acc;
    }

    // Row max across the 16-lane N group (xor of bits 0..3 stays in-group).
#pragma unroll
    for (int r = 0; r < 8; ++r) {
      float v = mtile[r];
      for (int off = 1; off < 16; off <<= 1)
        v = fmaxf(v, __shfl_xor(v, off, 32));
      mtile[r] = v;
    }

    // Online softmax update.
    float rscale[8];
#pragma unroll
    for (int r = 0; r < 8; ++r) {
      float mnew = fmaxf(mrow[r], mtile[r]);
      rscale[r]  = __expf(mrow[r] - mnew);
      mrow[r]    = mnew;
      lrow[r]   *= rscale[r];
    }
#pragma unroll
    for (int dt = 0; dt < 8; ++dt)
#pragma unroll
      for (int r = 0; r < 8; ++r) oacc[dt][r] *= rscale[r];

#pragma unroll
    for (int nt = 0; nt < 4; ++nt)
#pragma unroll
      for (int r = 0; r < 8; ++r) {
        float p = __expf(sacc[nt][r] - mrow[r]);
        lrow[r] += p;                          // per-lane partial; reduced at end
        Pt[m0 + r + halfSel * 8][nt * 16 + lanelo] = (_Float16)p;
      }
    __syncthreads();

    // O += P * V  (A = Pt rows, B = VT rows, K-dim = 64 keys in 2 chunks of 32)
#pragma unroll
    for (int dt = 0; dt < 8; ++dt) {
#pragma unroll
      for (int kk = 0; kk < 2; ++kk) {
        const _Float16* prow = &Pt[m0 + lanelo][kk * 32];
        v16h a = cat8(*(const v8h*)(prow + halfSel * 16),
                      *(const v8h*)(prow + halfSel * 16 + 8));
        const _Float16* vtrow = &VT[dt * 16 + lanelo][kk * 32 + halfSel * 16];
        v16h b = cat8(*(const v8h*)vtrow, *(const v8h*)(vtrow + 8));
        oacc[dt] = __builtin_amdgcn_wmma_f32_16x16x32_f16(false, a, false, b,
                                                          (short)0, oacc[dt], false, false);
      }
    }
    __syncthreads();
  }

  // Reduce row sums across the 16-lane group, normalize, and write out
  // back to [s, h*128+d] layout (f16) for the output projection GEMM.
#pragma unroll
  for (int r = 0; r < 8; ++r) {
    float v = lrow[r];
    for (int off = 1; off < 16; off <<= 1) v += __shfl_xor(v, off, 32);
    lrow[r] = 1.0f / v;
  }
#pragma unroll
  for (int dt = 0; dt < 8; ++dt)
#pragma unroll
    for (int r = 0; r < 8; ++r) {
      int row = qb + m0 + r + halfSel * 8;
      int d   = dt * 16 + lanelo;
      oh[(size_t)row * DMODEL + h * HD + d] = (_Float16)(oacc[dt][r] * lrow[r]);
    }
}

// ---------------------------------------------------------------------------
// Host-side orchestration (graph-capture safe: launches only).
// ---------------------------------------------------------------------------
extern "C" void kernel_launch(void* const* d_in, const int* in_sizes, int n_in,
                              void* d_out, int out_size, void* d_ws, size_t ws_size,
                              hipStream_t stream) {
  const float* x     = (const float*)d_in[0];
  const float* wq    = (const float*)d_in[1];
  const float* wk    = (const float*)d_in[2];
  const float* wv    = (const float*)d_in[3];
  const float* wo    = (const float*)d_in[4];
  const float* freqs = (const float*)d_in[5];
  // d_in[6] = start_pos (0 for this prefill configuration)

  char* ws = (char*)d_ws;
  _Float16* xh  = (_Float16*)(ws + 0);           // [3072][4096]      25.2 MB
  _Float16* wqh = (_Float16*)(ws + 25165824);    // [4096][4096]      33.6 MB
  _Float16* wkh = (_Float16*)(ws + 58720256);    // [1024][4096]       8.4 MB
  _Float16* wvh = (_Float16*)(ws + 67108864);    // [1024][4096]       8.4 MB
  _Float16* woh = (_Float16*)(ws + 75497472);    // [4096][4096]      33.6 MB
  float*    q32 = (float*)   (ws + 109051904);   // [3072][4096]      50.3 MB
  float*    k32 = (float*)   (ws + 159383552);   // [3072][1024]      12.6 MB
  float*    v32 = (float*)   (ws + 171966464);   // [3072][1024]      12.6 MB
  _Float16* qh  = (_Float16*)(ws + 184549376);   // [32][3072][128]   25.2 MB
  _Float16* kh  = (_Float16*)(ws + 209715200);   // [8][3072][128]     6.3 MB
  _Float16* vh  = (_Float16*)(ws + 216006656);   // [8][3072][128]     6.3 MB
  _Float16* oh  = (_Float16*)(ws + 222298112);   // [3072][4096]      25.2 MB

  auto cvt = [&](const float* s, _Float16* d, int n) {
    f32_to_f16<<<(n + 255) / 256, 256, 0, stream>>>(s, d, n);
  };
  cvt(x,  xh,  S_LEN * DMODEL);
  cvt(wq, wqh, DMODEL * DMODEL);
  cvt(wk, wkh, NKV * HD * DMODEL);
  cvt(wv, wvh, NKV * HD * DMODEL);
  cvt(wo, woh, DMODEL * DMODEL);

  // Projections: C = A * B^T
  gemm_nt_f16<<<dim3(S_LEN / 128, DMODEL / 128), 256, 0, stream>>>(xh, wqh, q32,
                                                                   S_LEN, DMODEL, DMODEL);
  gemm_nt_f16<<<dim3(S_LEN / 128, (NKV * HD) / 128), 256, 0, stream>>>(xh, wkh, k32,
                                                                       S_LEN, NKV * HD, DMODEL);
  gemm_nt_f16<<<dim3(S_LEN / 128, (NKV * HD) / 128), 256, 0, stream>>>(xh, wvh, v32,
                                                                       S_LEN, NKV * HD, DMODEL);

  // RoPE + head-major f16 re-layout
  {
    int nq = S_LEN * NH * 64;
    rope_to_heads<<<(nq + 255) / 256, 256, 0, stream>>>(q32, freqs, qh, NH);
    int nk = S_LEN * NKV * 64;
    rope_to_heads<<<(nk + 255) / 256, 256, 0, stream>>>(k32, freqs, kh, NKV);
    int nv = S_LEN * NKV * HD;
    v_to_heads<<<(nv + 255) / 256, 256, 0, stream>>>(v32, vh);
  }

  // Flash attention over the 2048-entry sliding-window cache
  flash_attn<<<dim3(S_LEN / QT, NH), 128, 0, stream>>>(qh, kh, vh, oh);

  // Output projection: d_out = oh * wo^T
  gemm_nt_f16<<<dim3(S_LEN / 128, DMODEL / 128), 256, 0, stream>>>(oh, woh, (float*)d_out,
                                                                   S_LEN, DMODEL, DMODEL);
}
